// EdgeAwareAttentionLayer_32229434589334
// MI455X (gfx1250) — matrix operand order repaired and checked by reading, not compiled, not run on hardware
//
#include <hip/hip_runtime.h>

// EdgeAwareAttentionLayer for MI455X (gfx1250), wave32 + WMMA f16->f32.
// Algebraic refactor: projections pushed before gathers, W_edge folded into
// Wk/Wv; node tensors (h,Q,K,V ~102MB) stay L2-resident (192MB L2).
// All weights pre-packed into per-lane WMMA B-fragment layout so each
// fragment load is one contiguous 32B vector load (2x global_load_b128).

typedef __attribute__((ext_vector_type(16))) _Float16 v16h;
typedef __attribute__((ext_vector_type(8)))  float    v8f;

#define NN 50000
#define EE 600000
#define D_IN 256
#define D 128

__device__ inline v8f wmma16x16x32(v16h a, v16h b, v8f c) {
  return __builtin_amdgcn_wmma_f32_16x16x32_f16(false, a, false, b, (short)0, c,
                                                false, false);
}

// A-matrix 16x32 f16 layout (ISA 7.12.2): lanes 0-15 row=lane, K 0..7 then 16..23;
// lanes 16-31 row=lane-16, K 8..15 then 24..31.  Two contiguous 16B chunks.
__device__ inline v16h load_a16x32(const _Float16* __restrict__ base, int ld, int lane) {
  const _Float16* p = base + (long)(lane & 15) * ld + ((lane >> 4) & 1) * 8;
  v16h a;
#pragma unroll
  for (int i = 0; i < 8; ++i) a[i] = p[i];
#pragma unroll
  for (int i = 0; i < 8; ++i) a[8 + i] = p[16 + i];
  return a;
}

// Pre-packed B fragment: one aligned 32B vector load per lane.
__device__ inline v16h load_b_packed(const _Float16* __restrict__ packed,
                                     int tile, int lane) {
  return ((const v16h*)packed)[tile * 32 + lane];
}

// Pack a row-major [Kdim x 128] f32 weight into WMMA B-fragment order:
// out[((nt*KT + kt)*32 + lane)*16 + i] with h=lane>>4: K=kt*32+8h+i (i<8),
// K=kt*32+16+8h+(i-8) (i>=8); col=nt*16+(lane&15).
__device__ inline void pack_b(const float* __restrict__ W, _Float16* __restrict__ out,
                              int Kdim, int tid) {
  int KT = Kdim >> 5;
  int total = Kdim * D;
  for (int idx = tid; idx < total; idx += 256) {
    int i = idx & 15;
    int lane = (idx >> 4) & 31;
    int t = idx >> 9;              // nt*KT + kt
    int kt = t % KT, nt = t / KT;
    int col = nt * 16 + (lane & 15);
    int klocal = ((i >> 3) << 4) + ((lane >> 4) << 3) + (i & 7);
    out[idx] = (_Float16)W[(kt * 32 + klocal) * D + col];
  }
}

__device__ inline float red16(float v) {  // reduce within 16-lane half-wave
  v += __shfl_xor(v, 1, 32);
  v += __shfl_xor(v, 2, 32);
  v += __shfl_xor(v, 4, 32);
  v += __shfl_xor(v, 8, 32);
  return v;
}

// order-preserving float<->uint for atomicMax
__device__ inline unsigned enc_f32(float f) {
  unsigned u = __float_as_uint(f);
  return (u & 0x80000000u) ? ~u : (u | 0x80000000u);
}
__device__ inline float dec_f32(unsigned u) {
  return __uint_as_float((u & 0x80000000u) ? (u & 0x7fffffffu) : ~u);
}

__global__ void f32_to_f16_kernel(const float* __restrict__ in,
                                  _Float16* __restrict__ out, long n) {
  long i = (long)blockIdx.x * blockDim.x + threadIdx.x;
  long stride = (long)gridDim.x * blockDim.x;
  for (; i < n; i += stride) out[i] = (_Float16)in[i];
}

// Fold edge weights through Wk/Wv; pack all GEMM weights into B-fragment layout.
__global__ __launch_bounds__(256) void prep_weights_kernel(
    const float* __restrict__ Wn, const float* __restrict__ Wq,
    const float* __restrict__ Wk, const float* __restrict__ Wv,
    const float* __restrict__ W1, const float* __restrict__ W2,
    const float* __restrict__ We, const float* __restrict__ be,
    _Float16* Wn_p, _Float16* Wq_p, _Float16* Wk_p, _Float16* Wv_p,
    _Float16* W1_p, _Float16* W2_p, _Float16* Wek_p, _Float16* Wev_p,
    float* bek, float* bev) {
  __shared__ float Wekf[32 * D];
  __shared__ float Wevf[32 * D];
  int tid = threadIdx.x;
  pack_b(Wn, Wn_p, D_IN, tid);
  pack_b(Wq, Wq_p, D, tid);
  pack_b(Wk, Wk_p, D, tid);
  pack_b(Wv, Wv_p, D, tid);
  pack_b(W1, W1_p, D, tid);
  pack_b(W2, W2_p, D, tid);
  for (int o = tid; o < 32 * D; o += 256) {  // Wek = W_edge@Wk, Wev = W_edge@Wv
    int d = o >> 7, n = o & 127;
    float sk = 0.f, sv = 0.f;
    for (int m = 0; m < D; ++m) {
      float w = We[d * D + m];
      sk += w * Wk[m * D + n];
      sv += w * Wv[m * D + n];
    }
    Wekf[o] = sk; Wevf[o] = sv;
  }
  for (int n = tid; n < D; n += 256) {       // bek = b_edge@Wk, bev = b_edge@Wv
    float sk = 0.f, sv = 0.f;
    for (int m = 0; m < D; ++m) { sk += be[m] * Wk[m * D + n]; sv += be[m] * Wv[m * D + n]; }
    bek[n] = sk; bev[n] = sv;
  }
  __syncthreads();
  pack_b(Wekf, Wek_p, 32, tid);
  pack_b(Wevf, Wev_p, 32, tid);
}

// Fused: h = x@Wn + bn ; Q=h@Wq ; K=h@Wk ; V=h@Wv.  16 rows/block, 8 waves,
// each wave owns 16 of the 128 output columns. 20 WMMAs per wave.
__global__ __launch_bounds__(256) void node_qkv_kernel(
    const _Float16* __restrict__ xh, const _Float16* __restrict__ Wn_p,
    const float* __restrict__ bn, const _Float16* __restrict__ Wq_p,
    const _Float16* __restrict__ Wk_p, const _Float16* __restrict__ Wv_p,
    float* __restrict__ h, float* __restrict__ Q, float* __restrict__ K,
    float* __restrict__ V) {
  __shared__ _Float16 hh_s[16 * D];
  int tid = threadIdx.x, lane = tid & 31, w = tid >> 5;
  long row0 = (long)blockIdx.x * 16;
  v8f c = {};
#pragma unroll
  for (int kt = 0; kt < 8; ++kt) {           // K = 256
    v16h a = load_a16x32(xh + row0 * D_IN + kt * 32, D_IN, lane);
    v16h b = load_b_packed(Wn_p, w * 8 + kt, lane);
    c = wmma16x16x32(a, b, c);
  }
  int col = w * 16 + (lane & 15);
  int rbase = (lane < 16) ? 0 : 8;
  float bias = bn[col];
#pragma unroll
  for (int i = 0; i < 8; ++i) {
    float v = c[i] + bias;
    h[(row0 + rbase + i) * D + col] = v;
    hh_s[(rbase + i) * D + col] = (_Float16)v;
  }
  __syncthreads();
  v8f cq = {}, ck = {}, cv = {};
#pragma unroll
  for (int kt = 0; kt < 4; ++kt) {           // K = 128
    v16h a = load_a16x32(hh_s + kt * 32, D, lane);
    v16h bq = load_b_packed(Wq_p, w * 4 + kt, lane);
    v16h bk = load_b_packed(Wk_p, w * 4 + kt, lane);
    v16h bv = load_b_packed(Wv_p, w * 4 + kt, lane);
    cq = wmma16x16x32(a, bq, cq);
    ck = wmma16x16x32(a, bk, ck);
    cv = wmma16x16x32(a, bv, cv);
  }
#pragma unroll
  for (int i = 0; i < 8; ++i) {
    long r = row0 + rbase + i;
    Q[r * D + col] = cq[i]; K[r * D + col] = ck[i]; V[r * D + col] = cv[i];
  }
}

// Per 16-edge tile: K_edge via one WMMA (K dim == 32 exactly), add bek and
// L2-resident K[src], dot with Q[dst], scale -> logits; block max -> atomicMax.
__global__ __launch_bounds__(256) void edge_attn_kernel(
    const _Float16* __restrict__ eah, const long long* __restrict__ ei,
    const _Float16* __restrict__ Wek_p, const float* __restrict__ bek,
    const float* __restrict__ Knode, const float* __restrict__ Qnode,
    float* __restrict__ attn, unsigned* __restrict__ gmax_enc) {
  __shared__ float k_tile[16][D];
  __shared__ float sattn[16];
  int tid = threadIdx.x, lane = tid & 31, w = tid >> 5;
  long row0 = (long)blockIdx.x * 16;
  v16h a = load_a16x32(eah + row0 * 32, 32, lane);
  v16h b = load_b_packed(Wek_p, w, lane);
  v8f c = {};
  c = wmma16x16x32(a, b, c);
  int col = w * 16 + (lane & 15);
  int rbase = (lane < 16) ? 0 : 8;
  float bias = bek[col];
#pragma unroll
  for (int i = 0; i < 8; ++i) k_tile[rbase + i][col] = c[i] + bias;
  __syncthreads();
  int j = tid >> 4;
  long src = (long)ei[row0 + j];
  long dst = (long)ei[(long)EE + row0 + j];
  const float* Kn = Knode + src * D;
  const float* Qd = Qnode + dst * D;
  float part = 0.f;
#pragma unroll
  for (int i = 0; i < 8; ++i) {
    int cc = (tid & 15) + i * 16;
    part += Qd[cc] * (k_tile[j][cc] + Kn[cc]);
  }
  part = red16(part);
  if ((tid & 15) == 0) {
    float av = part * 0.0883883476483184f;  // 1/sqrt(128)
    attn[row0 + j] = av;
    sattn[j] = av;
  }
  __syncthreads();
  if (tid == 0) {
    float m = sattn[0];
    for (int i = 1; i < 16; ++i) m = fmaxf(m, sattn[i]);
    atomicMax(gmax_enc, enc_f32(m));
  }
}

__global__ void edge_softmax_kernel(const long long* __restrict__ ei,
                                    float* __restrict__ attn,
                                    float* __restrict__ denom,
                                    const unsigned* __restrict__ gmax_enc) {
  float gmax = dec_f32(*gmax_enc);
  long i = (long)blockIdx.x * blockDim.x + threadIdx.x;
  long stride = (long)gridDim.x * blockDim.x;
  for (; i < EE; i += stride) {
    float wv = expf(attn[i] - gmax);
    attn[i] = wv;
    atomicAdd(denom + (long)ei[(long)EE + i], wv);
  }
}

// Recompute V_edge by WMMA (cheaper than a 307MB round-trip), scatter alpha*v.
__global__ __launch_bounds__(256) void edge_aggregate_kernel(
    const _Float16* __restrict__ eah, const long long* __restrict__ ei,
    const _Float16* __restrict__ Wev_p, const float* __restrict__ bev,
    const float* __restrict__ Vnode, const float* __restrict__ attnw,
    const float* __restrict__ denom, float* __restrict__ out_accum) {
  __shared__ float v_tile[16][D];
  int tid = threadIdx.x, lane = tid & 31, w = tid >> 5;
  long row0 = (long)blockIdx.x * 16;
  v16h a = load_a16x32(eah + row0 * 32, 32, lane);
  v16h b = load_b_packed(Wev_p, w, lane);
  v8f c = {};
  c = wmma16x16x32(a, b, c);
  int col = w * 16 + (lane & 15);
  int rbase = (lane < 16) ? 0 : 8;
  float bias = bev[col];
#pragma unroll
  for (int i = 0; i < 8; ++i) v_tile[rbase + i][col] = c[i] + bias;
  __syncthreads();
  int j = tid >> 4;
  long src = (long)ei[row0 + j];
  long dst = (long)ei[(long)EE + row0 + j];
  float alpha = attnw[row0 + j] / (denom[dst] + 1e-6f);
  const float* Vn = Vnode + src * D;
  float* op = out_accum + dst * D;
#pragma unroll
  for (int i = 0; i < 8; ++i) {
    int cc = (tid & 15) + i * 16;
    atomicAdd(op + cc, alpha * (v_tile[j][cc] + Vn[cc]));
  }
}

// LN -> MLP(W1,gelu,W2) via WMMA -> LN. 16 rows/block, 16 lanes per row for LN.
__global__ __launch_bounds__(256) void node_post_kernel(
    const float* __restrict__ h, const float* __restrict__ out_accum,
    const _Float16* __restrict__ W1_p, const float* __restrict__ b1,
    const _Float16* __restrict__ W2_p, const float* __restrict__ b2,
    const float* __restrict__ gamma, const float* __restrict__ beta,
    float* __restrict__ out) {
  __shared__ float zf[16][D];
  __shared__ _Float16 yh[16 * D];
  __shared__ _Float16 mh[16 * D];
  int tid = threadIdx.x, lane = tid & 31, w = tid >> 5;
  long row0 = (long)blockIdx.x * 16;
  int j = tid >> 4, c0 = tid & 15;
  long base = (row0 + j) * D;

  // LN1 over (out_accum + h)
  float vals[8], s = 0.f;
#pragma unroll
  for (int i = 0; i < 8; ++i) {
    int cc = c0 + i * 16;
    float v = out_accum[base + cc] + h[base + cc];
    vals[i] = v; s += v;
  }
  s = red16(s);
  float mu = s * (1.f / D);
  float s2 = 0.f;
#pragma unroll
  for (int i = 0; i < 8; ++i) { float d = vals[i] - mu; s2 += d * d; }
  s2 = red16(s2);
  float rstd = rsqrtf(s2 * (1.f / D) + 1e-5f);
#pragma unroll
  for (int i = 0; i < 8; ++i) {
    int cc = c0 + i * 16;
    float y = (vals[i] - mu) * rstd * gamma[cc] + beta[cc];
    zf[j][cc] = y;
    yh[j * D + cc] = (_Float16)y;
  }
  __syncthreads();

  // MLP layer 1 + exact gelu
  v8f c1 = {};
#pragma unroll
  for (int kt = 0; kt < 4; ++kt) {
    v16h a = load_a16x32(yh + kt * 32, D, lane);
    v16h b = load_b_packed(W1_p, w * 4 + kt, lane);
    c1 = wmma16x16x32(a, b, c1);
  }
  int col = w * 16 + (lane & 15);
  int rbase = (lane < 16) ? 0 : 8;
#pragma unroll
  for (int i = 0; i < 8; ++i) {
    float v = c1[i] + b1[col];
    v = 0.5f * v * (1.f + erff(v * 0.70710678118f));
    mh[(rbase + i) * D + col] = (_Float16)v;
  }
  __syncthreads();

  // MLP layer 2 + residual
  v8f c2 = {};
#pragma unroll
  for (int kt = 0; kt < 4; ++kt) {
    v16h a = load_a16x32(mh + kt * 32, D, lane);
    v16h b = load_b_packed(W2_p, w * 4 + kt, lane);
    c2 = wmma16x16x32(a, b, c2);
  }
  float rv[8];
#pragma unroll
  for (int i = 0; i < 8; ++i) rv[i] = c2[i] + b2[col] + zf[rbase + i][col];
  __syncthreads();
#pragma unroll
  for (int i = 0; i < 8; ++i) zf[rbase + i][col] = rv[i];
  __syncthreads();

  // LN2
  s = 0.f;
#pragma unroll
  for (int i = 0; i < 8; ++i) { vals[i] = zf[j][c0 + i * 16]; s += vals[i]; }
  s = red16(s);
  mu = s * (1.f / D);
  s2 = 0.f;
#pragma unroll
  for (int i = 0; i < 8; ++i) { float d = vals[i] - mu; s2 += d * d; }
  s2 = red16(s2);
  rstd = rsqrtf(s2 * (1.f / D) + 1e-5f);
#pragma unroll
  for (int i = 0; i < 8; ++i) {
    int cc = c0 + i * 16;
    out[base + cc] = (vals[i] - mu) * rstd * gamma[cc] + beta[cc];
  }
}

extern "C" void kernel_launch(void* const* d_in, const int* in_sizes, int n_in,
                              void* d_out, int out_size, void* d_ws, size_t ws_size,
                              hipStream_t stream) {
  (void)in_sizes; (void)n_in; (void)out_size; (void)ws_size;
  const float*     x      = (const float*)d_in[0];
  const long long* ei     = (const long long*)d_in[1];   // int64 [2,E]
  const float*     ea     = (const float*)d_in[2];
  const float*     W_node = (const float*)d_in[3];
  const float*     b_node = (const float*)d_in[4];
  const float*     W_edge = (const float*)d_in[5];
  const float*     b_edge = (const float*)d_in[6];
  const float*     Wq     = (const float*)d_in[7];
  const float*     Wk     = (const float*)d_in[8];
  const float*     Wv     = (const float*)d_in[9];
  const float*     W1     = (const float*)d_in[10];
  const float*     b1     = (const float*)d_in[11];
  const float*     W2     = (const float*)d_in[12];
  const float*     b2     = (const float*)d_in[13];
  const float*     gamma  = (const float*)d_in[14];
  const float*     beta   = (const float*)d_in[15];
  float* out = (float*)d_out;

  char* ws = (char*)d_ws;
  size_t off = 0;
  auto alloc = [&](size_t bytes) -> void* {
    void* p = ws + off;
    off = (off + bytes + 255) & ~(size_t)255;
    return p;
  };
  _Float16* xh    = (_Float16*)alloc((size_t)NN * D_IN * 2);
  _Float16* eah   = (_Float16*)alloc((size_t)EE * 32 * 2);
  float*    h     = (float*)alloc((size_t)NN * D * 4);
  float*    Q     = (float*)alloc((size_t)NN * D * 4);
  float*    K     = (float*)alloc((size_t)NN * D * 4);
  float*    V     = (float*)alloc((size_t)NN * D * 4);
  float*    attn  = (float*)alloc((size_t)EE * 4);
  float*    denom = (float*)alloc((size_t)NN * 4);
  float*    oacc  = (float*)alloc((size_t)NN * D * 4);
  _Float16* Wn_p  = (_Float16*)alloc((size_t)D_IN * D * 2);
  _Float16* Wq_p  = (_Float16*)alloc((size_t)D * D * 2);
  _Float16* Wk_p  = (_Float16*)alloc((size_t)D * D * 2);
  _Float16* Wv_p  = (_Float16*)alloc((size_t)D * D * 2);
  _Float16* W1_p  = (_Float16*)alloc((size_t)D * D * 2);
  _Float16* W2_p  = (_Float16*)alloc((size_t)D * D * 2);
  _Float16* Wek_p = (_Float16*)alloc((size_t)32 * D * 2);
  _Float16* Wev_p = (_Float16*)alloc((size_t)32 * D * 2);
  float*    bek   = (float*)alloc((size_t)D * 4);
  float*    bev   = (float*)alloc((size_t)D * 4);
  unsigned* gmax  = (unsigned*)alloc(4);

  hipMemsetAsync(denom, 0, (size_t)NN * 4, stream);
  hipMemsetAsync(oacc, 0, (size_t)NN * D * 4, stream);
  hipMemsetAsync(gmax, 0, 4, stream);  // enc(-NaN): identity for encoded max

  f32_to_f16_kernel<<<8192, 256, 0, stream>>>(x, xh, (long)NN * D_IN);
  f32_to_f16_kernel<<<8192, 256, 0, stream>>>(ea, eah, (long)EE * 32);
  prep_weights_kernel<<<1, 256, 0, stream>>>(W_node, Wq, Wk, Wv, W1, W2, W_edge,
                                             b_edge, Wn_p, Wq_p, Wk_p, Wv_p,
                                             W1_p, W2_p, Wek_p, Wev_p, bek, bev);
  node_qkv_kernel<<<NN / 16, 256, 0, stream>>>(xh, Wn_p, b_node, Wq_p, Wk_p,
                                               Wv_p, h, Q, K, V);
  edge_attn_kernel<<<EE / 16, 256, 0, stream>>>(eah, ei, Wek_p, bek, K, Q, attn,
                                                gmax);
  edge_softmax_kernel<<<2344, 256, 0, stream>>>(ei, attn, denom, gmax);
  edge_aggregate_kernel<<<EE / 16, 256, 0, stream>>>(eah, ei, Wev_p, bev, V,
                                                     attn, denom, oacc);
  node_post_kernel<<<NN / 16, 256, 0, stream>>>(h, oacc, W1_p, b1, W2_p, b2,
                                                gamma, beta, out);
}
